// TransConv_77197742178786
// MI455X (gfx1250) — matrix operand order repaired and checked by reading, not compiled, not run on hardware
//
#include <hip/hip_runtime.h>
#include <hip/hip_bf16.h>

// ---------------------------------------------------------------------------
// SGFormer-style TransConv for MI455X (gfx1250)
//   - bf16 activations in HBM (halves dominant traffic; GEMM math was bf16)
//   - async global->LDS tile staging (global_load_async_to_lds_b128, ASYNCcnt)
//   - wave32 bf16 WMMA, fp32 accumulation; fp32 stats/LN/residual/output
//   - sched_barrier pins all fragment ds_load_b128s ahead of the WMMA burst
// ---------------------------------------------------------------------------

typedef __attribute__((ext_vector_type(16))) __bf16       v16bf;
typedef __attribute__((ext_vector_type(8)))  float        v8f;
typedef __attribute__((ext_vector_type(4)))  unsigned int v4u;

struct u4x2 { v4u lo; v4u hi; };

static __device__ __forceinline__ v16bf make_frag(const v4u* p, int i0, int i1)
{
    u4x2 t;
    t.lo = p[i0];
    t.hi = p[i1];
    return __builtin_bit_cast(v16bf, t);
}

// exact bf16 -> fp32 (bit tricks; no cvt needed)
static __device__ __forceinline__ float bflo(unsigned u) { return __builtin_bit_cast(float, u << 16); }
static __device__ __forceinline__ float bfhi(unsigned u) { return __builtin_bit_cast(float, u & 0xffff0000u); }

// CDNA5 async global->LDS copy (16B per lane), tracked by ASYNCcnt
static __device__ __forceinline__ void async_ld_b128(unsigned ldsAddr, const void* g)
{
    asm volatile("global_load_async_to_lds_b128 %0, %1, off"
                 :: "v"(ldsAddr), "v"(g) : "memory");
}
static __device__ __forceinline__ void wait_async0()
{
    asm volatile("s_wait_asynccnt 0x0" ::: "memory");
}

#define NV_    20000
#define NC_    12768
#define NTOK   32768          // tokens per batch  (== 1<<15)
#define BATCH  4
#define MROWS  (BATCH * NTOK) // 131072 rows total
#define HID_   128
#define ALPHA_ 0.8f
#define EPSLN  1e-5f

// ---------------------------------------------------------------------------
// embed:  x = relu(LN(concat(vf@ve_w+ve_b, cf@ce_w+ce_b)));  prev = x  (bf16)
// ---------------------------------------------------------------------------
__global__ __launch_bounds__(256)
void embed_kernel(const float* __restrict__ vf, const float* __restrict__ cf,
                  const float* __restrict__ vew, const float* __restrict__ veb,
                  const float* __restrict__ cew, const float* __restrict__ ceb,
                  const float* __restrict__ g,   const float* __restrict__ bt,
                  __bf16* __restrict__ X, __bf16* __restrict__ PREV)
{
    const int wid  = threadIdx.x >> 5;
    const int lane = threadIdx.x & 31;
    const int row  = blockIdx.x * 8 + wid;         // 0..MROWS-1
    const int b    = row >> 15;
    const int n    = row & (NTOK - 1);

    const float* in; const float* W; const float* bi; int Kd;
    if (n < NV_) { in = vf + ((size_t)b * NV_ + n) * 18;          W = vew; bi = veb; Kd = 18; }
    else         { in = cf + ((size_t)b * NC_ + (n - NV_)) * 5;   W = cew; bi = ceb; Kd = 5;  }

    float acc[4];
    #pragma unroll
    for (int j = 0; j < 4; ++j) acc[j] = bi[lane + 32 * j];
    for (int k = 0; k < Kd; ++k) {
        float f = in[k];
        #pragma unroll
        for (int j = 0; j < 4; ++j) acc[j] = fmaf(f, W[k * HID_ + lane + 32 * j], acc[j]);
    }
    float s1 = 0.f, s2 = 0.f;
    #pragma unroll
    for (int j = 0; j < 4; ++j) { s1 += acc[j]; s2 += acc[j] * acc[j]; }
    #pragma unroll
    for (int m = 1; m < 32; m <<= 1) { s1 += __shfl_xor(s1, m, 32); s2 += __shfl_xor(s2, m, 32); }
    float mean = s1 * (1.f / HID_);
    float var  = s2 * (1.f / HID_) - mean * mean;
    float rs   = rsqrtf(var + EPSLN);
    #pragma unroll
    for (int j = 0; j < 4; ++j) {
        int d = lane + 32 * j;
        float o = fmaxf((acc[j] - mean) * rs * g[d] + bt[d], 0.f);
        size_t oi = (size_t)row * HID_ + d;
        __bf16 ob = (__bf16)o;
        X[oi] = ob; PREV[oi] = ob;
    }
}

// ---------------------------------------------------------------------------
// Generic bf16-WMMA GEMM: out[M,NOUT] = epilogue( X[M,K] @ W[K,NOUT] )
//   Xin is bf16 in HBM -> async-copied (b128/lane) into padded LDS tile.
//   sW: [NOUT][40] bf16, column-transposed 32-K chunk (B frags = 2x b128)
//   EPI 0: +bias raw (QKV) | 1: relu (FFN1) | 2: attention+mix+LN | 3: mix+LN
// ---------------------------------------------------------------------------
template<int K, int NOUT, int EPI, typename OT>
__global__ __launch_bounds__(256)
void gemm_kernel(const __bf16* __restrict__ Xin, const float* __restrict__ Wg,
                 const float* __restrict__ bias, OT* __restrict__ out,
                 const __bf16* __restrict__ vbuf, const __bf16* __restrict__ resbuf,
                 const float* __restrict__ normbuf, const float* __restrict__ scal,
                 const float* __restrict__ gamma, const float* __restrict__ beta,
                 int wPerBatch)
{
    constexpr int NT  = NOUT / 32;     // 16-wide n-tiles per wave
    constexpr int XS  = K + 8;         // sX row stride (bf16); (XS*2)%16==0
    constexpr int XS4 = XS / 8;        // sX row stride (v4u)
    constexpr int WS  = 40;            // sW row stride (bf16)
    constexpr int SEG = K / 8;         // 16-byte segments per activation row
    extern __shared__ char smem[];
    __bf16* sW = (__bf16*)smem;                      // NOUT x 40
    __bf16* sX = (__bf16*)(smem + NOUT * WS * 2);    // 64 x XS
    float*  sRed = (float*)smem;                     // reused after final barrier

    const int tid  = threadIdx.x;
    const int lane = tid & 31, wid = tid >> 5;
    const int n0   = lane & 15, hlf = lane >> 4;
    const int mwarp = wid & 3,  nwarp = wid >> 2;
    const int row0 = blockIdx.x * 64;
    const float* Wp = Wg + (size_t)(row0 >> 15) * (size_t)wPerBatch;

    // async-stage activation tile (raw bf16 bytes, padded row stride)
    {
        unsigned sxAddr = (unsigned)(size_t)sX;
        for (int idx = tid; idx < 64 * SEG; idx += 256) {
            int r = idx / SEG, s = idx - r * SEG;
            async_ld_b128(sxAddr + (unsigned)((r * XS + s * 8) * 2),
                          Xin + (size_t)(row0 + r) * K + s * 8);
        }
        wait_async0();
    }

    v8f acc[NT];
    v8f z8 = 0.f;
    #pragma unroll
    for (int nt = 0; nt < NT; ++nt) acc[nt] = z8;

    const v4u* sX4 = (const v4u*)sX;
    const v4u* sW4 = (const v4u*)sW;

    #pragma unroll
    for (int ks = 0; ks < K / 32; ++ks) {
        __syncthreads();
        // stage 32 x NOUT weight chunk, transposed per column: sW[n][kl]
        for (int idx = tid; idx < 32 * NOUT; idx += 256) {
            int kl = idx / NOUT, nn = idx - kl * NOUT;
            sW[nn * WS + kl] = (__bf16)Wp[(size_t)(ks * 32 + kl) * NOUT + nn];
        }
        __syncthreads();

        // batch ALL fragment loads for this K-step, then a single wait + burst
        int abase = (mwarp * 16 + n0) * XS4 + ks * 4 + hlf;
        v16bf afrag = make_frag(sX4, abase, abase + 2);
        v16bf bfrag[NT];
        #pragma unroll
        for (int nt = 0; nt < NT; ++nt) {
            int nc = nwarp * (NOUT / 2) + nt * 16 + n0;
            int bbase = nc * (WS / 8) + hlf;           // WS/8 = 5 v4u per column
            bfrag[nt] = make_frag(sW4, bbase, bbase + 2);
        }
        // pin loads ahead of the WMMA burst (no instruction may cross)
        __builtin_amdgcn_sched_barrier(0);
        #pragma unroll
        for (int nt = 0; nt < NT; ++nt) {
            acc[nt] = __builtin_amdgcn_wmma_f32_16x16x32_bf16(
                false, afrag, false, bfrag[nt], (short)0, acc[nt], false, false);
        }
    }
    __syncthreads();

    if constexpr (EPI == 0) {
        #pragma unroll
        for (int nt = 0; nt < NT; ++nt) {
            int col = nwarp * (NOUT / 2) + nt * 16 + n0;
            float bv = bias ? bias[col] : 0.f;
            #pragma unroll
            for (int r = 0; r < 8; ++r) {
                int gRow = row0 + mwarp * 16 + hlf * 8 + r;
                out[(size_t)gRow * NOUT + col] = (OT)(acc[nt][r] + bv);
            }
        }
    } else if constexpr (EPI == 1) {
        #pragma unroll
        for (int nt = 0; nt < NT; ++nt) {
            int col = nwarp * (NOUT / 2) + nt * 16 + n0;
            #pragma unroll
            for (int r = 0; r < 8; ++r) {
                int gRow = row0 + mwarp * 16 + hlf * 8 + r;
                out[(size_t)gRow * NOUT + col] = (OT)fmaxf(acc[nt][r], 0.f);
            }
        }
    } else {
        // EPI 2 / 3 : mix + LayerNorm (NOUT == 128)
        float mixv[NT * 8];
        float scale = 0.f;
        if constexpr (EPI == 2) scale = rsqrtf(scal[0]) * rsqrtf(scal[1]);
        #pragma unroll
        for (int r = 0; r < 8; ++r) {
            int gRow = row0 + mwarp * 16 + hlf * 8 + r;
            float den = 1.f;
            if constexpr (EPI == 2) den = normbuf[gRow] * scale + (float)NTOK;
            float s1 = 0.f, s2 = 0.f;
            #pragma unroll
            for (int nt = 0; nt < NT; ++nt) {
                int col = nwarp * (NOUT / 2) + nt * 16 + n0;
                size_t gi = (size_t)gRow * NOUT + col;
                float m;
                if constexpr (EPI == 2) {
                    float a = acc[nt][r] * scale + (float)NTOK * (float)vbuf[gi];
                    m = ALPHA_ * (a / den) + (1.f - ALPHA_) * (float)resbuf[gi];
                } else {
                    m = ALPHA_ * acc[nt][r] + (1.f - ALPHA_) * (float)resbuf[gi];
                }
                mixv[nt * 8 + r] = m;
                s1 += m; s2 += m * m;
            }
            #pragma unroll
            for (int msk = 1; msk < 16; msk <<= 1) {
                s1 += __shfl_xor(s1, msk, 32);
                s2 += __shfl_xor(s2, msk, 32);
            }
            if (n0 == 0) {
                int rr = mwarp * 16 + hlf * 8 + r;
                sRed[rr * 4 + nwarp * 2 + 0] = s1;
                sRed[rr * 4 + nwarp * 2 + 1] = s2;
            }
        }
        __syncthreads();
        #pragma unroll
        for (int r = 0; r < 8; ++r) {
            int rr = mwarp * 16 + hlf * 8 + r;
            int gRow = row0 + rr;
            float s1 = sRed[rr * 4 + 0] + sRed[rr * 4 + 2];
            float s2 = sRed[rr * 4 + 1] + sRed[rr * 4 + 3];
            float mean = s1 * (1.f / NOUT);
            float var  = s2 * (1.f / NOUT) - mean * mean;
            float rs   = rsqrtf(var + EPSLN);
            #pragma unroll
            for (int nt = 0; nt < NT; ++nt) {
                int col = nwarp * (NOUT / 2) + nt * 16 + n0;
                float o = (mixv[nt * 8 + r] - mean) * rs * gamma[col] + beta[col];
                out[(size_t)gRow * NOUT + col] = (OT)o;
            }
        }
    }
}

// ---------------------------------------------------------------------------
// stats over q,k : Sum(q^2), Sum(k^2) (whole-tensor Frobenius), colsum(k) per b
// bf16 inputs read as packed uints; exact shift-conversion to fp32
// ---------------------------------------------------------------------------
__global__ __launch_bounds__(256)
void qk_stats_kernel(const __bf16* __restrict__ Q, const __bf16* __restrict__ Kb,
                     float* __restrict__ ksum, float* __restrict__ scal)
{
    const unsigned* Qu = (const unsigned*)Q;    // 64 uints per row
    const unsigned* Ku = (const unsigned*)Kb;
    const int row0 = blockIdx.x * 256;
    const int b = row0 >> 15;
    const int dp  = threadIdx.x & 63;           // d-pair index (0..63)
    const int sub = threadIdx.x >> 6;           // 0..3
    float sq = 0.f, sk = 0.f, ks0 = 0.f, ks1 = 0.f;
    for (int r = sub; r < 256; r += 4) {
        size_t idx = (size_t)(row0 + r) * 64 + dp;
        unsigned qd = Qu[idx], kd = Ku[idx];
        float q0 = bflo(qd), q1 = bfhi(qd);
        float k0 = bflo(kd), k1 = bfhi(kd);
        sq += q0 * q0 + q1 * q1;
        sk += k0 * k0 + k1 * k1;
        ks0 += k0; ks1 += k1;
    }
    __shared__ float red[256];
    red[threadIdx.x] = ks0; __syncthreads();
    if (sub == 0) atomicAdd(&ksum[b * HID_ + 2 * dp],
                            ks0 + red[64 + dp] + red[128 + dp] + red[192 + dp]);
    __syncthreads();
    red[threadIdx.x] = ks1; __syncthreads();
    if (sub == 0) atomicAdd(&ksum[b * HID_ + 2 * dp + 1],
                            ks1 + red[64 + dp] + red[128 + dp] + red[192 + dp]);
    __syncthreads();
    red[threadIdx.x] = sq; __syncthreads();
    for (int s = 128; s > 0; s >>= 1) { if (threadIdx.x < s) red[threadIdx.x] += red[threadIdx.x + s]; __syncthreads(); }
    if (threadIdx.x == 0) atomicAdd(&scal[0], red[0]);
    __syncthreads();
    red[threadIdx.x] = sk; __syncthreads();
    for (int s = 128; s > 0; s >>= 1) { if (threadIdx.x < s) red[threadIdx.x] += red[threadIdx.x + s]; __syncthreads(); }
    if (threadIdx.x == 0) atomicAdd(&scal[1], red[0]);
}

// ---------------------------------------------------------------------------
// kvs[b] = k[b]^T @ v[b]  (128x128 per batch), WMMA reduction over N
// manual staging (transpose during store); fragments are b128 loads
// ---------------------------------------------------------------------------
__global__ __launch_bounds__(256)
void kv_kernel(const __bf16* __restrict__ Kb, const __bf16* __restrict__ Vb,
               float* __restrict__ kvs)
{
    extern __shared__ char smem[];
    __bf16* sKT = (__bf16*)smem;                   // [128][72]  k^T (A operand)
    __bf16* sV  = (__bf16*)(smem + 128 * 72 * 2);  // [128][72]  v^T (B operand)
    const int b  = blockIdx.x >> 5;
    const int c0 = blockIdx.x & 31;
    const int tid = threadIdx.x, lane = tid & 31, wid = tid >> 5;
    const int n0 = lane & 15, hlf = lane >> 4;

    v8f acc[8];
    v8f z8 = 0.f;
    #pragma unroll
    for (int nt = 0; nt < 8; ++nt) acc[nt] = z8;

    const v4u* sKT4 = (const v4u*)sKT;   // 9 v4u per row
    const v4u* sV4  = (const v4u*)sV;
    const unsigned* Ku = (const unsigned*)Kb;
    const unsigned* Vu = (const unsigned*)Vb;

    for (int chunk = c0 * 64; chunk < NTOK; chunk += 32 * 64) {
        size_t baseu = ((size_t)b * NTOK + chunk) * 64;     // uint index
        for (int idx = tid; idx < 64 * 64; idx += 256) {
            int r = idx >> 6, cp = idx & 63;                // r = n, cp = d-pair
            unsigned kd = Ku[baseu + (size_t)r * 64 + cp];
            unsigned vd = Vu[baseu + (size_t)r * 64 + cp];
            int c = cp * 2;
            sKT[c * 72 + r]       = __builtin_bit_cast(__bf16, (unsigned short)(kd & 0xffff));
            sKT[(c + 1) * 72 + r] = __builtin_bit_cast(__bf16, (unsigned short)(kd >> 16));
            sV [c * 72 + r]       = __builtin_bit_cast(__bf16, (unsigned short)(vd & 0xffff));
            sV [(c + 1) * 72 + r] = __builtin_bit_cast(__bf16, (unsigned short)(vd >> 16));
        }
        __syncthreads();
        #pragma unroll
        for (int ks = 0; ks < 2; ++ks) {
            // batch loads: A frag + all 8 B frags, then WMMA burst
            int abase = (wid * 16 + n0) * 9 + ks * 4 + hlf;
            v16bf afrag = make_frag(sKT4, abase, abase + 2);
            v16bf bfrag[8];
            #pragma unroll
            for (int nt = 0; nt < 8; ++nt) {
                int nc = nt * 16 + n0;
                int bbase = nc * 9 + ks * 4 + hlf;
                bfrag[nt] = make_frag(sV4, bbase, bbase + 2);
            }
            __builtin_amdgcn_sched_barrier(0);
            #pragma unroll
            for (int nt = 0; nt < 8; ++nt) {
                acc[nt] = __builtin_amdgcn_wmma_f32_16x16x32_bf16(
                    false, afrag, false, bfrag[nt], (short)0, acc[nt], false, false);
            }
        }
        __syncthreads();
    }
    float* outp = kvs + (size_t)b * HID_ * HID_;
    #pragma unroll
    for (int nt = 0; nt < 8; ++nt) {
        int col = nt * 16 + n0;
        #pragma unroll
        for (int r = 0; r < 8; ++r) {
            int row = wid * 16 + hlf * 8 + r;
            atomicAdd(&outp[row * HID_ + col], acc[nt][r]);
        }
    }
}

// ---------------------------------------------------------------------------
// normalizer[row] = dot(q[row,:], ks_sum[b,:])   (raw; scaled in attn epilogue)
// ---------------------------------------------------------------------------
__global__ __launch_bounds__(256)
void normalizer_kernel(const __bf16* __restrict__ Q, const float* __restrict__ ksum,
                       float* __restrict__ norm)
{
    const unsigned* Qu = (const unsigned*)Q;   // 64 uints per row
    const int wid = threadIdx.x >> 5, lane = threadIdx.x & 31;
    const int row = blockIdx.x * 8 + wid;
    const int b = row >> 15;
    unsigned u0 = Qu[(size_t)row * 64 + lane];
    unsigned u1 = Qu[(size_t)row * 64 + 32 + lane];
    const float* kp = ksum + b * HID_;
    float s = bflo(u0) * kp[2 * lane]      + bfhi(u0) * kp[2 * lane + 1]
            + bflo(u1) * kp[64 + 2 * lane] + bfhi(u1) * kp[64 + 2 * lane + 1];
    #pragma unroll
    for (int m = 1; m < 32; m <<= 1) s += __shfl_xor(s, m, 32);
    if (lane == 0) norm[row] = s;
}

__global__ void zero_kernel(float* __restrict__ p, int n)
{
    int i = blockIdx.x * blockDim.x + threadIdx.x;
    if (i < n) p[i] = 0.f;
}

// ---------------------------------------------------------------------------
extern "C" void kernel_launch(void* const* d_in, const int* in_sizes, int n_in,
                              void* d_out, int out_size, void* d_ws, size_t ws_size,
                              hipStream_t stream)
{
    const float* cfeat = (const float*)d_in[0];
    const float* vfeat = (const float*)d_in[3];
    const float* vew  = (const float*)d_in[5];
    const float* veb  = (const float*)d_in[6];
    const float* cew  = (const float*)d_in[7];
    const float* ceb  = (const float*)d_in[8];
    const float* lng  = (const float*)d_in[9];
    const float* lnb  = (const float*)d_in[10];
    const float* Wq   = (const float*)d_in[11];
    const float* Wqb  = (const float*)d_in[12];
    const float* Wk   = (const float*)d_in[13];
    const float* Wkb  = (const float*)d_in[14];
    const float* Wv   = (const float*)d_in[15];
    const float* Wvb  = (const float*)d_in[16];
    const float* fc1  = (const float*)d_in[17];
    const float* fc2  = (const float*)d_in[18];

    char* wsb = (char*)d_ws;
    const size_t SZB = (size_t)MROWS * HID_ * sizeof(__bf16);   // 33.5MB per stream
    __bf16* X    = (__bf16*)(wsb + 0 * SZB);
    __bf16* PREV = (__bf16*)(wsb + 1 * SZB);
    __bf16* Q    = (__bf16*)(wsb + 2 * SZB);
    __bf16* Kbuf = (__bf16*)(wsb + 3 * SZB);
    __bf16* V    = (__bf16*)(wsb + 4 * SZB);
    __bf16* H    = Q;                       // FFN hidden [M,256] aliases dead q/k
    float*  KVS  = (float*)(wsb + 5 * SZB);
    float*  KSUM = KVS + BATCH * HID_ * HID_;
    float*  SCAL = KSUM + BATCH * HID_;
    float*  NORM = SCAL + 16;

    dim3 blk(256);
    const int gemmBlocks = MROWS / 64;            // 2048
    const int lds_128_128 = 128 * 40 * 2 + 64 * (128 + 8) * 2;   // 27.6 KB
    const int lds_128_256 = 256 * 40 * 2 + 64 * (128 + 8) * 2;   // 37.9 KB
    const int lds_256_128 = 128 * 40 * 2 + 64 * (256 + 8) * 2;   // 44.0 KB
    const int lds_kv      = 2 * 128 * 72 * 2;                    // 36.9 KB

    embed_kernel<<<MROWS / 8, blk, 0, stream>>>(vfeat, cfeat, vew, veb, cew, ceb,
                                                lng, lnb, X, PREV);

    for (int i = 0; i < 2; ++i) {
        const int nz = BATCH * HID_ * HID_ + BATCH * HID_ + 16;
        zero_kernel<<<(nz + 255) / 256, blk, 0, stream>>>(KVS, nz);

        gemm_kernel<128, 128, 0, __bf16><<<gemmBlocks, blk, lds_128_128, stream>>>(
            X, Wq + (size_t)i * HID_ * HID_, Wqb + i * HID_, Q,
            nullptr, nullptr, nullptr, nullptr, nullptr, nullptr, 0);
        gemm_kernel<128, 128, 0, __bf16><<<gemmBlocks, blk, lds_128_128, stream>>>(
            X, Wk + (size_t)i * HID_ * HID_, Wkb + i * HID_, Kbuf,
            nullptr, nullptr, nullptr, nullptr, nullptr, nullptr, 0);
        gemm_kernel<128, 128, 0, __bf16><<<gemmBlocks, blk, lds_128_128, stream>>>(
            X, Wv + (size_t)i * HID_ * HID_, Wvb + i * HID_, V,
            nullptr, nullptr, nullptr, nullptr, nullptr, nullptr, 0);

        qk_stats_kernel<<<512, blk, 0, stream>>>(Q, Kbuf, KSUM, SCAL);
        kv_kernel<<<BATCH * 32, blk, lds_kv, stream>>>(Kbuf, V, KVS);
        normalizer_kernel<<<MROWS / 8, blk, 0, stream>>>(Q, KSUM, NORM);

        // attention: (q@kvs)*s + N*v, /(q.ks_sum*s + N), alpha-mix w/ prev, LN
        gemm_kernel<128, 128, 2, __bf16><<<gemmBlocks, blk, lds_128_128, stream>>>(
            Q, KVS, nullptr, X, V, PREV, NORM, SCAL,
            lng + (2 * i + 1) * HID_, lnb + (2 * i + 1) * HID_, HID_ * HID_);

        // FFN
        gemm_kernel<128, 256, 1, __bf16><<<gemmBlocks, blk, lds_128_256, stream>>>(
            X, fc1 + (size_t)i * HID_ * 2 * HID_, nullptr, H,
            nullptr, nullptr, nullptr, nullptr, nullptr, nullptr, 0);
        if (i == 0) {
            gemm_kernel<256, 128, 3, __bf16><<<gemmBlocks, blk, lds_256_128, stream>>>(
                H, fc2 + (size_t)i * 2 * HID_ * HID_, nullptr, X,
                nullptr, X, nullptr, nullptr,
                lng + (2 * i + 2) * HID_, lnb + (2 * i + 2) * HID_, 0);
            hipMemcpyAsync(PREV, X, SZB, hipMemcpyDeviceToDevice, stream);
        } else {
            gemm_kernel<256, 128, 3, float><<<gemmBlocks, blk, lds_256_128, stream>>>(
                H, fc2 + (size_t)i * 2 * HID_ * HID_, nullptr, (float*)d_out,
                nullptr, X, nullptr, nullptr,
                lng + (2 * i + 2) * HID_, lnb + (2 * i + 2) * HID_, 0);
        }
    }
    (void)in_sizes; (void)n_in; (void)out_size; (void)ws_size;
}
